// SparseAttention_42528766165017
// MI455X (gfx1250) — compile-verified, matrix operand order checked by reading, not run on hardware
//
#include <hip/hip_runtime.h>
#include <hip/hip_bf16.h>

// ---- sizes (match reference) ----
#define E_   1024
#define H_   16
#define D_   64
#define B_   2
#define N_   2048
#define M_   (B_ * N_)   // 4096 tokens

typedef __attribute__((ext_vector_type(16))) _Float16 v16h;
typedef __attribute__((ext_vector_type(8)))  _Float16 v8h;
typedef __attribute__((ext_vector_type(8)))  float    v8f;
typedef __attribute__((ext_vector_type(4)))  unsigned v4u;
typedef __attribute__((ext_vector_type(8)))  int      v8i;
typedef __attribute__((ext_vector_type(4)))  int      v4i;

union V16U { v16h v; v8h h[2]; };

// Low 32 bits of a flat shared-memory pointer are the LDS byte offset
__device__ __forceinline__ unsigned sa_lds_off(const void* p) {
    return (unsigned)(unsigned long long)(uintptr_t)p;
}

// TDM: DMA a 2D f16 tile (tile_w x tile_h elements, row stride d0_stride
// elements in global memory) into LDS at byte offset lds_off.
// Descriptor bitfields per CDNA5 ISA 8.3/8.4: group0 {count=1, lds_addr,
// global_addr, type=2}; group1 {data_size=2B, tensor dims = tile dims,
// tile dims, dim0 stride}; groups 2/3 zero (2D tensor).
__device__ __forceinline__ void sa_tdm_load_f16(unsigned lds_off,
                                                const _Float16* gptr,
                                                unsigned d0_stride,
                                                unsigned tile_w,
                                                unsigned tile_h) {
    unsigned long long ga = (unsigned long long)(uintptr_t)gptr;
    v4u g0 = { 1u,                                   // count=1
               lds_off,                              // lds_addr
               (unsigned)ga,                         // global_addr[31:0]
               ((unsigned)(ga >> 32) & 0x01FFFFFFu) | 0x80000000u }; // addr[56:32] | type=2
    v8i g1 = { (int)0x00010000u,                     // wg_mask=0, data_size=1 (2B)
               (int)((tile_w & 0xFFFFu) << 16),      // tensor_dim0 lo16
               (int)(((tile_w >> 16) & 0xFFFFu) | ((tile_h & 0xFFFFu) << 16)), // d0 hi / d1 lo
               (int)(((tile_h >> 16) & 0xFFFFu) | ((tile_w & 0xFFFFu) << 16)), // d1 hi / tile_dim0
               (int)(tile_h & 0xFFFFu),              // tile_dim1 (tile_dim2=0)
               (int)d0_stride,                       // tensor_dim0_stride lo32
               0, 0 };
    v4i gz = { 0, 0, 0, 0 };
#if __clang_major__ >= 23
    v8i gz8 = { 0, 0, 0, 0, 0, 0, 0, 0 };
    __builtin_amdgcn_tensor_load_to_lds(g0, g1, gz, gz, gz8, 0);
#else
    __builtin_amdgcn_tensor_load_to_lds(g0, g1, gz, gz, 0);
#endif
}

// ---------------------------------------------------------------------------
// Prep kernels
// ---------------------------------------------------------------------------

__global__ __launch_bounds__(256) void sa_cvt_x(const float* __restrict__ x,
                                                _Float16* __restrict__ xh) {
    int idx = blockIdx.x * 256 + threadIdx.x;
    xh[idx] = (_Float16)x[idx];
}

//  mode 0:  out[n,k] = sum_j W[k, h*64+j] * Wp[j, i]   (n = h*64+i)   [fold Wp]
//  mode 1:  out[n,k] = W[k,n]                                          [plain T]
__global__ __launch_bounds__(256) void sa_fuse_w(const float* __restrict__ W,
                                                 const float* __restrict__ Wp,
                                                 _Float16* __restrict__ outT,
                                                 int mode) {
    int idx = blockIdx.x * 256 + threadIdx.x;
    int n = idx >> 10, k = idx & 1023;
    if (mode == 0) {
        int hh = n >> 6, i = n & 63;
        float acc = 0.f;
        #pragma unroll 8
        for (int j = 0; j < 64; ++j)
            acc += W[k * E_ + hh * 64 + j] * Wp[j * 64 + i];
        outT[(size_t)n * E_ + k] = (_Float16)acc;
    } else {
        outT[(size_t)n * E_ + k] = (_Float16)W[(size_t)k * E_ + n];
    }
}

__global__ __launch_bounds__(256) void sa_fuse_b(const float* __restrict__ bsrc,
                                                 const float* __restrict__ Wp,
                                                 float* __restrict__ bdst,
                                                 int mode) {
    int n = blockIdx.x * 256 + threadIdx.x;
    if (n >= E_) return;
    if (mode == 0) {
        int hh = n >> 6, i = n & 63;
        float acc = 0.f;
        for (int j = 0; j < 64; ++j) acc += bsrc[hh * 64 + j] * Wp[j * 64 + i];
        bdst[n] = acc;
    } else {
        bdst[n] = bsrc[n];
    }
}

// ---------------------------------------------------------------------------
// QKV GEMM with TDM-staged, double-buffered LDS tiles.
// Block tile 128(M) x 64(N); 256 threads = 8 waves; wave w owns rows w*16..+16.
// ---------------------------------------------------------------------------
__global__ __launch_bounds__(256) void sa_qkv_gemm(const _Float16* __restrict__ xh,
                                                   const _Float16* __restrict__ Wt,
                                                   const float* __restrict__ bias,
                                                   _Float16* __restrict__ out,
                                                   int transposeOut) {
    __shared__ _Float16 AsBuf[2][128 * 32];   // 2 x 8KB
    __shared__ _Float16 BsBuf[2][64 * 32];    // 2 x 4KB

    const int m0 = blockIdx.x * 128;
    const int n0 = blockIdx.y * 64;
    const int t = threadIdx.x;
    const int w = t >> 5;
    const int lane = t & 31;
    const int lrow = lane & 15;
    const int lhalf = lane >> 4;

    v8f c[4] = {};

    // preload k-chunk 0 into buffer 0 (one TDM pair, issued by wave 0)
    if (w == 0) {
        sa_tdm_load_f16(sa_lds_off(&AsBuf[0][0]), xh + (size_t)m0 * E_, E_, 32, 128);
        sa_tdm_load_f16(sa_lds_off(&BsBuf[0][0]), Wt + (size_t)n0 * E_, E_, 32, 64);
    }

    int curb = 0;
    for (int kc = 0; kc < E_; kc += 32) {
        if (w == 0) {
            if (kc + 32 < E_) {   // prefetch next pair into other buffer
                sa_tdm_load_f16(sa_lds_off(&AsBuf[curb ^ 1][0]),
                                xh + (size_t)m0 * E_ + kc + 32, E_, 32, 128);
                sa_tdm_load_f16(sa_lds_off(&BsBuf[curb ^ 1][0]),
                                Wt + (size_t)n0 * E_ + kc + 32, E_, 32, 64);
                __builtin_amdgcn_s_wait_tensorcnt(2);   // current pair done
            } else {
                __builtin_amdgcn_s_wait_tensorcnt(0);
            }
        }
        __syncthreads();

        const _Float16* As = AsBuf[curb];
        const _Float16* Bs = BsBuf[curb];

        V16U a;
        int arow = w * 16 + lrow;
        a.h[0] = *(const v8h*)&As[arow * 32 + lhalf * 8];
        a.h[1] = *(const v8h*)&As[arow * 32 + 16 + lhalf * 8];

        #pragma unroll
        for (int nt = 0; nt < 4; ++nt) {
            V16U bfr;
            int brow = nt * 16 + lrow;
            bfr.h[0] = *(const v8h*)&Bs[brow * 32 + lhalf * 8];
            bfr.h[1] = *(const v8h*)&Bs[brow * 32 + 16 + lhalf * 8];
            c[nt] = __builtin_amdgcn_wmma_f32_16x16x32_f16(
                false, a.v, false, bfr.v, (short)0, c[nt], false, false);
        }
        __syncthreads();   // all waves done reading buf before TDM reuses it
        curb ^= 1;
    }

    #pragma unroll
    for (int nt = 0; nt < 4; ++nt) {
        int nf = n0 + nt * 16 + lrow;
        float bval = bias[nf];
        int hh = nf >> 6, dd = nf & 63;
        #pragma unroll
        for (int r = 0; r < 8; ++r) {
            int mrow = m0 + w * 16 + r + lhalf * 8;
            int bidx = mrow >> 11;
            int tok = mrow & (N_ - 1);
            float val = c[nt][r] + bval;
            size_t addr;
            if (!transposeOut)
                addr = ((size_t)(bidx * H_ + hh) * N_ + tok) * D_ + dd;
            else
                addr = ((size_t)(bidx * H_ + hh) * D_ + dd) * N_ + tok;
            out[addr] = (_Float16)val;
        }
    }
}

// ---------------------------------------------------------------------------
// Flash attention. 4 waves/block share one (b,h): K/V tiles are TDM-staged
// into double-buffered LDS once per 32-key block and shared by all 4 waves.
// ---------------------------------------------------------------------------
__global__ __launch_bounds__(128) void sa_attn(const _Float16* __restrict__ Q,
                                               const _Float16* __restrict__ K,
                                               const _Float16* __restrict__ Vt,
                                               float* __restrict__ out) {
    __shared__ _Float16 KsB[2][32 * 64];     // 2 x 4KB, K tile [key][d]
    __shared__ _Float16 VsB[2][64 * 32];     // 2 x 4KB, V tile [d][key]
    __shared__ _Float16 Pl4[4][16 * 32];     // per-wave P slice

    const int w = threadIdx.x >> 5;
    const int lane = threadIdx.x & 31;
    const int lrow = lane & 15;
    const int lhalf = lane >> 4;

    const int wid = blockIdx.x * 4 + w;      // 4 consecutive wids share (b,h)
    const int qt = wid & 127;
    const int h  = (wid >> 7) & 15;
    const int b  = wid >> 11;

    const _Float16* Qp = Q  + ((size_t)(b * H_ + h) * N_ + qt * 16) * D_;
    const _Float16* Kp = K  +  (size_t)(b * H_ + h) * N_ * D_;
    const _Float16* Vp = Vt +  (size_t)(b * H_ + h) * D_ * N_;
    _Float16* Pl = Pl4[w];

    // Q fragments (two K=32 chunks of the d=64 contraction)
    V16U aq[2];
    #pragma unroll
    for (int ch = 0; ch < 2; ++ch) {
        int dlo = ch * 32 + lhalf * 8;
        aq[ch].h[0] = *(const v8h*)&Qp[lrow * D_ + dlo];
        aq[ch].h[1] = *(const v8h*)&Qp[lrow * D_ + dlo + 16];
    }

    float mrun[8], lrun[8];
    v8f o[4] = {};
    #pragma unroll
    for (int r = 0; r < 8; ++r) { mrun[r] = -__builtin_inff(); lrun[r] = 0.f; }

    // preload key-block 0 (wave 0): K tile 32x64, V tile 64x32
    if (w == 0) {
        sa_tdm_load_f16(sa_lds_off(&KsB[0][0]), Kp, D_, 64, 32);
        sa_tdm_load_f16(sa_lds_off(&VsB[0][0]), Vp, N_, 32, 64);
    }

    int curb = 0;
    for (int kb = 0; kb < N_; kb += 32) {
        if (w == 0) {
            if (kb + 32 < N_) {
                sa_tdm_load_f16(sa_lds_off(&KsB[curb ^ 1][0]),
                                Kp + (size_t)(kb + 32) * D_, D_, 64, 32);
                sa_tdm_load_f16(sa_lds_off(&VsB[curb ^ 1][0]),
                                Vp + kb + 32, N_, 32, 64);
                __builtin_amdgcn_s_wait_tensorcnt(2);
            } else {
                __builtin_amdgcn_s_wait_tensorcnt(0);
            }
        }
        __syncthreads();

        const _Float16* Ks = KsB[curb];
        const _Float16* Vs = VsB[curb];

        // ---- S = Q K^T : two 16x16 key tiles ----
        v8f s[2];
        #pragma unroll
        for (int st = 0; st < 2; ++st) {
            v8f cc = {};
            #pragma unroll
            for (int ch = 0; ch < 2; ++ch) {
                V16U bk;
                const _Float16* kr = &Ks[(st * 16 + lrow) * D_ + ch * 32 + lhalf * 8];
                bk.h[0] = *(const v8h*)kr;
                bk.h[1] = *(const v8h*)(kr + 16);
                cc = __builtin_amdgcn_wmma_f32_16x16x32_f16(
                    false, aq[ch].v, false, bk.v, (short)0, cc, false, false);
            }
            s[st] = cc;
        }

        // ---- online softmax (rows: half0 -> r, half1 -> r+8) ----
        float p0[8], p1[8];
        #pragma unroll
        for (int r = 0; r < 8; ++r) {
            float s0 = s[0][r] * 0.125f;     // 1/sqrt(64)
            float s1 = s[1][r] * 0.125f;
            float vm = fmaxf(s0, s1);
            #pragma unroll
            for (int off = 1; off < 16; off <<= 1)
                vm = fmaxf(vm, __shfl_xor(vm, off, 32));
            float mnew  = fmaxf(mrun[r], vm);
            float alpha = __expf(mrun[r] - mnew);
            float e0 = __expf(s0 - mnew);
            float e1 = __expf(s1 - mnew);
            float rs = e0 + e1;
            #pragma unroll
            for (int off = 1; off < 16; off <<= 1)
                rs += __shfl_xor(rs, off, 32);
            lrun[r] = lrun[r] * alpha + rs;
            mrun[r] = mnew;
            #pragma unroll
            for (int nt = 0; nt < 4; ++nt) o[nt][r] *= alpha;
            p0[r] = e0; p1[r] = e1;
        }

        // ---- P (C layout) -> LDS 16x32, reload as A fragment ----
        #pragma unroll
        for (int r = 0; r < 8; ++r) {
            int row = r + lhalf * 8;
            Pl[row * 32 + lrow]      = (_Float16)p0[r];
            Pl[row * 32 + 16 + lrow] = (_Float16)p1[r];
        }
        V16U ap;
        ap.h[0] = *(const v8h*)&Pl[lrow * 32 + lhalf * 8];
        ap.h[1] = *(const v8h*)&Pl[lrow * 32 + 16 + lhalf * 8];

        // ---- O += P @ V (V tile already transposed: [d][key]) ----
        #pragma unroll
        for (int nt = 0; nt < 4; ++nt) {
            V16U bv;
            const _Float16* vr = &Vs[(nt * 16 + lrow) * 32 + lhalf * 8];
            bv.h[0] = *(const v8h*)vr;
            bv.h[1] = *(const v8h*)(vr + 16);
            o[nt] = __builtin_amdgcn_wmma_f32_16x16x32_f16(
                false, ap.v, false, bv.v, (short)0, o[nt], false, false);
        }
        __syncthreads();   // done reading K/V bufs before TDM overwrites
        curb ^= 1;
    }

    // ---- normalize and store fp32 output [b, n, h*64 + d] ----
    #pragma unroll
    for (int r = 0; r < 8; ++r) {
        float inv = 1.f / lrun[r];
        int row = qt * 16 + r + lhalf * 8;
        #pragma unroll
        for (int nt = 0; nt < 4; ++nt) {
            out[((size_t)b * N_ + row) * E_ + h * 64 + nt * 16 + lrow] =
                o[nt][r] * inv;
        }
    }
}

// ---------------------------------------------------------------------------
// Host launcher
// ---------------------------------------------------------------------------
extern "C" void kernel_launch(void* const* d_in, const int* in_sizes, int n_in,
                              void* d_out, int out_size, void* d_ws, size_t ws_size,
                              hipStream_t stream) {
    const float* x  = (const float*)d_in[0];
    const float* Wq = (const float*)d_in[1];
    const float* bq = (const float*)d_in[2];
    const float* Wk = (const float*)d_in[3];
    const float* bk = (const float*)d_in[4];
    const float* Wv = (const float*)d_in[5];
    const float* bv = (const float*)d_in[6];
    const float* Wp = (const float*)d_in[7];
    float* out = (float*)d_out;

    char* wsb = (char*)d_ws;
    _Float16* xh   = (_Float16*)(wsb);                         // 8 MB
    _Float16* Wq2t = (_Float16*)(wsb + (size_t)8  * 1048576);  // 2 MB
    _Float16* Wk2t = (_Float16*)(wsb + (size_t)10 * 1048576);  // 2 MB
    _Float16* Wvt  = (_Float16*)(wsb + (size_t)12 * 1048576);  // 2 MB
    float*    bq2  = (float*)   (wsb + (size_t)14 * 1048576);  // 4 KB
    float*    bk2  = bq2 + E_;
    float*    bv2  = bk2 + E_;
    _Float16* Qb   = (_Float16*)(wsb + (size_t)14 * 1048576 + 16384);  // 8 MB
    _Float16* Kb   = Qb + (size_t)B_ * H_ * N_ * D_;                    // 8 MB
    _Float16* Vtb  = Kb + (size_t)B_ * H_ * N_ * D_;                    // 8 MB

    sa_cvt_x<<<(M_ * E_) / 256, 256, 0, stream>>>(x, xh);

    sa_fuse_w<<<(E_ * E_) / 256, 256, 0, stream>>>(Wq, Wp, Wq2t, 0);
    sa_fuse_w<<<(E_ * E_) / 256, 256, 0, stream>>>(Wk, Wp, Wk2t, 0);
    sa_fuse_w<<<(E_ * E_) / 256, 256, 0, stream>>>(Wv, Wp, Wvt, 1);
    sa_fuse_b<<<4, 256, 0, stream>>>(bq, Wp, bq2, 0);
    sa_fuse_b<<<4, 256, 0, stream>>>(bk, Wp, bk2, 0);
    sa_fuse_b<<<4, 256, 0, stream>>>(bv, Wp, bv2, 1);

    dim3 ggrid(M_ / 128, E_ / 64);
    sa_qkv_gemm<<<ggrid, 256, 0, stream>>>(xh, Wq2t, bq2, Qb, 0);
    sa_qkv_gemm<<<ggrid, 256, 0, stream>>>(xh, Wk2t, bk2, Kb, 0);
    sa_qkv_gemm<<<ggrid, 256, 0, stream>>>(xh, Wvt,  bv2, Vtb, 1);

    sa_attn<<<(B_ * H_ * (N_ / 16)) / 4, 128, 0, stream>>>(Qb, Kb, Vtb, out);
}
